// mpnn_forward_model_44495861187263
// MI455X (gfx1250) — matrix operand order repaired and checked by reading, not compiled, not run on hardware
//
#include <hip/hip_runtime.h>
#include <hip/hip_bf16.h>

// ---------------------------------------------------------------------------
// GraphNet forward for MI455X (gfx1250, wave32, WMMA).
//   * all GEMMs via v_wmma_f32_16x16x32_bf16 (bf16 in, f32 accumulate)
//   * edge latents el kept in bf16 (halves the dominant HBM stream)
//   * el rows staged into LDS with GLOBAL_LOAD_ASYNC_TO_LDS_B128 (ASYNCcnt)
//   * gathers staged as float4x2 -> bf16x8 -> ds_store_b128
// ---------------------------------------------------------------------------

typedef __attribute__((ext_vector_type(16))) __bf16 v16bf;
typedef __attribute__((ext_vector_type(8)))  __bf16 bf16x8;
typedef __attribute__((ext_vector_type(2)))  __bf16 bf16x2;
typedef __attribute__((ext_vector_type(8)))  float  v8f;

static constexpr int Nn   = 50000;
static constexpr int Ee   = 800000;
static constexpr int NF   = 5;
static constexpr int EF   = 4;
static constexpr int Hh   = 64;
static constexpr int OUTD = 7;

#define USE_ASYNC_LDS_COPY 1

__device__ inline void async_copy_b128(void* lds_dst, const void* gsrc) {
#if USE_ASYNC_LDS_COPY
  // ISA aperture rule: LDS byte offset = flat_addr[31:0] for shared pointers.
  unsigned ldso = (unsigned)(size_t)lds_dst;
  asm volatile("global_load_async_to_lds_b128 %0, %1, off"
               :: "v"(ldso), "v"(gsrc) : "memory");
#else
  *(bf16x8*)lds_dst = *(const bf16x8*)gsrc;
#endif
}
__device__ inline void async_copy_wait() {
#if USE_ASYNC_LDS_COPY
  asm volatile("s_wait_asynccnt 0" ::: "memory");
#endif
}

__device__ inline v8f wmma_bf16(v16bf a, v16bf b, v8f c) {
  return __builtin_amdgcn_wmma_f32_16x16x32_bf16(
      false, a, false, b, (short)0, c, false, false);
}

// A fragment: 16x32 bf16 from LDS row-major [rows][ldm]; base at (row0,k0).
__device__ inline v16bf load_A_frag(const __bf16* base, int ldm, int lane) {
  const int m    = lane & 15;
  const int hofs = (lane >> 4) << 3;
  const __bf16* r = base + m * ldm;
  v16bf a;
#pragma unroll
  for (int v = 0; v < 8; ++v) {
    const int k = ((v < 4) ? (2 * v) : (2 * v + 8)) + hofs;
    bf16x2 p = *(const bf16x2*)(r + k);
    a[2 * v]     = p[0];
    a[2 * v + 1] = p[1];
  }
  return a;
}

// B fragment from pre-swizzled weights: lane-contiguous 32B load.
__device__ inline v16bf load_B_swz(const __bf16* __restrict__ W, int ntiles,
                                   int kt, int nt, int lane) {
  return *(const v16bf*)(W + (size_t)((kt * ntiles + nt) * 32 + lane) * 16);
}

__device__ inline bf16x8 pack8(float4 a, float4 b, float s) {
  bf16x8 o;
  o[0] = (__bf16)(a.x * s); o[1] = (__bf16)(a.y * s);
  o[2] = (__bf16)(a.z * s); o[3] = (__bf16)(a.w * s);
  o[4] = (__bf16)(b.x * s); o[5] = (__bf16)(b.y * s);
  o[6] = (__bf16)(b.z * s); o[7] = (__bf16)(b.w * s);
  return o;
}

// ---------------------------------------------------------------------------
// Utility kernels
// ---------------------------------------------------------------------------
__global__ void zero_f32_kernel(float* p, size_t n) {
  size_t i = (size_t)blockIdx.x * blockDim.x + threadIdx.x;
  size_t s = (size_t)gridDim.x * blockDim.x;
  for (; i < n; i += s) p[i] = 0.f;
}

__global__ void count_deg_kernel(const int* __restrict__ dst,
                                 float* __restrict__ cnt, int E) {
  int i = blockIdx.x * blockDim.x + threadIdx.x;
  if (i < E) atomicAdd(&cnt[dst[i]], 1.0f);
}

__global__ void inv_clip_kernel(float* c, int n) {
  int i = blockIdx.x * blockDim.x + threadIdx.x;
  if (i < n) {
    float v = c[i];
    c[i] = 1.0f / (v > 1.0f ? v : 1.0f);
  }
}

// f32 weights [rows][cols] -> WMMA B-fragment swizzled bf16, zero padded.
__global__ void pack_w_swz_kernel(const float* __restrict__ w,
                                  __bf16* __restrict__ o,
                                  int rows, int cols, int ktiles, int ntiles) {
  int idx = blockIdx.x * blockDim.x + threadIdx.x;
  int total = ktiles * ntiles * 512;
  if (idx >= total) return;
  int e2   = idx & 15;
  int lane = (idx >> 4) & 31;
  int tile = idx >> 9;
  int nt = tile % ntiles;
  int kt = tile / ntiles;
  int k = 32 * kt + 8 * (lane >> 4) + (e2 & 7) + ((e2 >> 3) << 4);
  int n = 16 * nt + (lane & 15);
  float v = (k < rows && n < cols) ? w[k * cols + n] : 0.0f;
  o[idx] = (__bf16)v;
}

// ---------------------------------------------------------------------------
// Edge MLP: e = relu(concat(x[src],x[dst],eattr)@W1+b1)@W2+b2
// scatter-adds e into agg[dst]; optionally writes e (or el+=e) to el_out (bf16).
// 64 edges / block, 4 waves, wave = one 16-row M tile.
// ---------------------------------------------------------------------------
template <int XDIM, int EADIM, bool EA_BF16, bool WRITE_E, bool RESID_E>
__global__ __launch_bounds__(128)
void edge_mlp_kernel(const float* __restrict__ xn,
                     const float* __restrict__ eattr_f32,
                     const __bf16* el_in,
                     const int* __restrict__ src, const int* __restrict__ dst,
                     const __bf16* __restrict__ W1, const float* __restrict__ b1,
                     const __bf16* __restrict__ W2, const float* __restrict__ b2,
                     __bf16* el_out, float* __restrict__ agg, int Etot) {
  constexpr int IN_K = ((2 * XDIM + EADIM + 31) / 32) * 32;
  __shared__ __bf16 Atile[64 * IN_K];
  __shared__ __bf16 Htile[64 * 64];
  __shared__ __bf16 Otile[64 * 64];
  __shared__ int s_src[64], s_dst[64];

  const int tid = threadIdx.x;
  const int e0  = blockIdx.x * 64;

  if (tid < 64) {
    int e = e0 + tid;
    s_src[tid] = (e < Etot) ? src[e] : 0;
    s_dst[tid] = (e < Etot) ? dst[e] : 0;
  }
  __syncthreads();

  if constexpr (XDIM == 64 && EADIM == 64 && EA_BF16) {
    // x[src] | x[dst] gathers: 8-col chunks, float4x2 -> bf16x8 -> b128 store.
    for (int t = tid; t < 64 * 16; t += 128) {
      int row = t >> 4, c = t & 15;
      int e = e0 + row;
      bf16x8 o = {};
      if (e < Etot) {
        const float* p = (c < 8) ? xn + (size_t)s_src[row] * 64 + 8 * c
                                 : xn + (size_t)s_dst[row] * 64 + 8 * (c - 8);
        float4 a0 = *(const float4*)p;
        float4 a1 = *(const float4*)(p + 4);
        o = pack8(a0, a1, 1.0f);
      }
      *(bf16x8*)(Atile + row * IN_K + 8 * c) = o;
    }
    // el rows (already bf16): async DMA straight into LDS, no VGPR round trip.
    for (int t = tid; t < 64 * 8; t += 128) {
      int row = t >> 3, c = t & 7;
      int e = e0 + row;
      __bf16* ldst = Atile + row * IN_K + 2 * XDIM + 8 * c;
      if (e < Etot) {
        async_copy_b128(ldst, el_in + (size_t)e * 64 + 8 * c);
      } else {
        bf16x8 z = {};
        *(bf16x8*)ldst = z;
      }
    }
    async_copy_wait();
  } else {
    // Encoder path (tiny K): scalar staging with zero padding.
    for (int idx = tid; idx < 64 * IN_K; idx += 128) {
      int row = idx / IN_K;
      int col = idx - row * IN_K;
      int e = e0 + row;
      float v = 0.0f;
      if (e < Etot) {
        if (col < XDIM)                   v = xn[(size_t)s_src[row] * XDIM + col];
        else if (col < 2 * XDIM)          v = xn[(size_t)s_dst[row] * XDIM + (col - XDIM)];
        else if (col < 2 * XDIM + EADIM)  v = eattr_f32[(size_t)e * EADIM + (col - 2 * XDIM)];
      }
      Atile[idx] = (__bf16)v;
    }
  }
  __syncthreads();

  const int lane = tid & 31;
  const int wv   = tid >> 5;
  const int nl   = lane & 15;
  const int mofs = (lane >> 4) << 3;

  // Layer 1: [64,IN_K]@[IN_K,64], bias+ReLU, restage bf16.
#pragma unroll
  for (int nt = 0; nt < 4; ++nt) {
    v8f acc = {};
#pragma unroll
    for (int kt = 0; kt < IN_K / 32; ++kt) {
      v16bf a = load_A_frag(Atile + (16 * wv) * IN_K + 32 * kt, IN_K, lane);
      v16bf b = load_B_swz(W1, 4, kt, nt, lane);
      acc = wmma_bf16(a, b, acc);
    }
    float bias = b1[16 * nt + nl];
#pragma unroll
    for (int r = 0; r < 8; ++r) {
      float h = acc[r] + bias;
      Htile[(16 * wv + mofs + r) * 64 + 16 * nt + nl] = (__bf16)(h > 0.f ? h : 0.f);
    }
  }
  __syncthreads();

  // Layer 2: [64,64]@[64,64]+bias -> atomics scatter + (optional) Otile.
#pragma unroll
  for (int nt = 0; nt < 4; ++nt) {
    v8f acc = {};
#pragma unroll
    for (int kt = 0; kt < 2; ++kt) {
      v16bf a = load_A_frag(Htile + (16 * wv) * 64 + 32 * kt, 64, lane);
      v16bf b = load_B_swz(W2, 4, kt, nt, lane);
      acc = wmma_bf16(a, b, acc);
    }
    float bias = b2[16 * nt + nl];
#pragma unroll
    for (int r = 0; r < 8; ++r) {
      int row = 16 * wv + mofs + r;
      int e = e0 + row;
      float v = acc[r] + bias;
      int feat = 16 * nt + nl;
      if (e < Etot) atomicAdd(&agg[(size_t)s_dst[row] * 64 + feat], v);
      if (WRITE_E) Otile[row * 64 + feat] = (__bf16)v;
    }
  }

  if (WRITE_E) {
    __syncthreads();
    // Coalesced bf16 writeback; residual source (old el) read back out of the
    // already-staged A tile instead of a second global pass.
    for (int t = tid; t < 64 * 8; t += 128) {
      int row = t >> 3, c = t & 7;
      int e = e0 + row;
      if (e < Etot) {
        bf16x8 nv = *(bf16x8*)(Otile + row * 64 + 8 * c);
        if (RESID_E) {
          bf16x8 old = *(bf16x8*)(Atile + row * IN_K + 2 * XDIM + 8 * c);
#pragma unroll
          for (int i = 0; i < 8; ++i)
            nv[i] = (__bf16)((float)nv[i] + (float)old[i]);
        }
        *(bf16x8*)(el_out + (size_t)e * 64 + 8 * c) = nv;
      }
    }
  }
}

// ---------------------------------------------------------------------------
// Node MLP: o = relu(concat(xn, agg/deg)@W1+b1)@W2+b2 (+ xn residual)
// ---------------------------------------------------------------------------
template <int XDIM, int NT_OUT, bool RESID>
__global__ __launch_bounds__(128)
void node_mlp_kernel(const float* xn,
                     const float* __restrict__ agg, const float* __restrict__ invc,
                     const __bf16* __restrict__ W1, const float* __restrict__ b1,
                     const __bf16* __restrict__ W2, const float* __restrict__ b2,
                     float* out, int odim, int Ntot) {
  constexpr int IN_K = ((XDIM + 64 + 31) / 32) * 32;
  __shared__ __bf16 Atile[64 * IN_K];
  __shared__ __bf16 Htile[64 * 64];
  __shared__ float s_inv[64];

  const int tid = threadIdx.x;
  const int n0  = blockIdx.x * 64;

  if (tid < 64) {
    int n = n0 + tid;
    s_inv[tid] = (n < Ntot) ? invc[n] : 0.f;
  }
  __syncthreads();

  if constexpr (XDIM == 64) {
    for (int t = tid; t < 64 * 16; t += 128) {
      int row = t >> 4, c = t & 15;
      int n = n0 + row;
      bf16x8 o = {};
      if (n < Ntot) {
        if (c < 8) {
          const float* p = xn + (size_t)n * 64 + 8 * c;
          o = pack8(*(const float4*)p, *(const float4*)(p + 4), 1.0f);
        } else {
          const float* p = agg + (size_t)n * 64 + 8 * (c - 8);
          o = pack8(*(const float4*)p, *(const float4*)(p + 4), s_inv[row]);
        }
      }
      *(bf16x8*)(Atile + row * IN_K + 8 * c) = o;
    }
  } else {
    for (int idx = tid; idx < 64 * IN_K; idx += 128) {
      int row = idx / IN_K;
      int col = idx - row * IN_K;
      int n = n0 + row;
      float v = 0.0f;
      if (n < Ntot) {
        if (col < XDIM)            v = xn[(size_t)n * XDIM + col];
        else if (col < XDIM + 64)  v = agg[(size_t)n * 64 + (col - XDIM)] * s_inv[row];
      }
      Atile[idx] = (__bf16)v;
    }
  }
  __syncthreads();

  const int lane = tid & 31;
  const int wv   = tid >> 5;
  const int nl   = lane & 15;
  const int mofs = (lane >> 4) << 3;

#pragma unroll
  for (int nt = 0; nt < 4; ++nt) {
    v8f acc = {};
#pragma unroll
    for (int kt = 0; kt < IN_K / 32; ++kt) {
      v16bf a = load_A_frag(Atile + (16 * wv) * IN_K + 32 * kt, IN_K, lane);
      v16bf b = load_B_swz(W1, 4, kt, nt, lane);
      acc = wmma_bf16(a, b, acc);
    }
    float bias = b1[16 * nt + nl];
#pragma unroll
    for (int r = 0; r < 8; ++r) {
      float h = acc[r] + bias;
      Htile[(16 * wv + mofs + r) * 64 + 16 * nt + nl] = (__bf16)(h > 0.f ? h : 0.f);
    }
  }
  __syncthreads();

#pragma unroll
  for (int nt = 0; nt < NT_OUT; ++nt) {
    v8f acc = {};
#pragma unroll
    for (int kt = 0; kt < 2; ++kt) {
      v16bf a = load_A_frag(Htile + (16 * wv) * 64 + 32 * kt, 64, lane);
      v16bf b = load_B_swz(W2, NT_OUT, kt, nt, lane);
      acc = wmma_bf16(a, b, acc);
    }
    int c = 16 * nt + nl;
    float bias = (c < odim) ? b2[c] : 0.f;
#pragma unroll
    for (int r = 0; r < 8; ++r) {
      int row = 16 * wv + mofs + r;
      int n = n0 + row;
      if (n < Ntot && c < odim) {
        float v = acc[r] + bias;
        if (RESID) v += xn[(size_t)n * 64 + c];
        out[(size_t)n * odim + c] = v;
      }
    }
  }
}

// ---------------------------------------------------------------------------
// Host launch
// ---------------------------------------------------------------------------
extern "C" void kernel_launch(void* const* d_in, const int* in_sizes, int n_in,
                              void* d_out, int out_size, void* d_ws, size_t ws_size,
                              hipStream_t stream) {
  const float* x    = (const float*)d_in[0];
  const int*   eidx = (const int*)d_in[1];
  const float* ea   = (const float*)d_in[2];
  const float* enc_e_w1 = (const float*)d_in[3];  const float* enc_e_b1 = (const float*)d_in[4];
  const float* enc_e_w2 = (const float*)d_in[5];  const float* enc_e_b2 = (const float*)d_in[6];
  const float* enc_n_w1 = (const float*)d_in[7];  const float* enc_n_b1 = (const float*)d_in[8];
  const float* enc_n_w2 = (const float*)d_in[9];  const float* enc_n_b2 = (const float*)d_in[10];
  const float* proc_e_w1 = (const float*)d_in[11]; const float* proc_e_b1 = (const float*)d_in[12];
  const float* proc_e_w2 = (const float*)d_in[13]; const float* proc_e_b2 = (const float*)d_in[14];
  const float* proc_n_w1 = (const float*)d_in[15]; const float* proc_n_b1 = (const float*)d_in[16];
  const float* proc_n_w2 = (const float*)d_in[17]; const float* proc_n_b2 = (const float*)d_in[18];
  const float* dec_e_w1 = (const float*)d_in[19]; const float* dec_e_b1 = (const float*)d_in[20];
  const float* dec_e_w2 = (const float*)d_in[21]; const float* dec_e_b2 = (const float*)d_in[22];
  const float* dec_n_w1 = (const float*)d_in[23]; const float* dec_n_b1 = (const float*)d_in[24];
  const float* dec_n_w2 = (const float*)d_in[25]; const float* dec_n_b2 = (const float*)d_in[26];

  const int* src = eidx;
  const int* dst = eidx + Ee;

  char* ws = (char*)d_ws;
  size_t off = 0;
  auto take = [&](size_t bytes) -> char* {
    char* p = ws + off;
    off += (bytes + 255) & ~(size_t)255;
    return p;
  };
  __bf16* el  = (__bf16*)take((size_t)Ee * 64 * 2);  // edge latents (bf16)
  float*  agg = (float*)take((size_t)Nn * 64 * 4);   // scatter accumulator
  float*  xl  = (float*)take((size_t)Nn * 64 * 4);   // node latents
  float*  cnt = (float*)take((size_t)Nn * 4);        // degree -> 1/deg

  auto takeW = [&](int kt, int nt) -> __bf16* {
    return (__bf16*)take((size_t)kt * nt * 512 * 2);
  };
  __bf16* w_ee1 = takeW(1, 4);  __bf16* w_ee2 = takeW(2, 4);
  __bf16* w_en1 = takeW(3, 4);  __bf16* w_en2 = takeW(2, 4);
  __bf16* w_pe1 = takeW(6, 4);  __bf16* w_pe2 = takeW(2, 4);
  __bf16* w_pn1 = takeW(4, 4);  __bf16* w_pn2 = takeW(2, 4);
  __bf16* w_de1 = takeW(6, 4);  __bf16* w_de2 = takeW(2, 4);
  __bf16* w_dn1 = takeW(4, 4);  __bf16* w_dn2 = takeW(2, 1);

  auto packw = [&](const float* w, __bf16* o, int rows, int cols, int kt, int nt) {
    int total = kt * nt * 512;
    pack_w_swz_kernel<<<(total + 255) / 256, 256, 0, stream>>>(w, o, rows, cols, kt, nt);
  };
  packw(enc_e_w1, w_ee1, 2 * NF + EF, Hh, 1, 4);
  packw(enc_e_w2, w_ee2, Hh, Hh, 2, 4);
  packw(enc_n_w1, w_en1, NF + Hh, Hh, 3, 4);
  packw(enc_n_w2, w_en2, Hh, Hh, 2, 4);
  packw(proc_e_w1, w_pe1, 3 * Hh, Hh, 6, 4);
  packw(proc_e_w2, w_pe2, Hh, Hh, 2, 4);
  packw(proc_n_w1, w_pn1, 2 * Hh, Hh, 4, 4);
  packw(proc_n_w2, w_pn2, Hh, Hh, 2, 4);
  packw(dec_e_w1, w_de1, 3 * Hh, Hh, 6, 4);
  packw(dec_e_w2, w_de2, Hh, Hh, 2, 4);
  packw(dec_n_w1, w_dn1, 2 * Hh, Hh, 4, 4);
  packw(dec_n_w2, w_dn2, Hh, OUTD, 2, 1);

  zero_f32_kernel<<<256, 256, 0, stream>>>(cnt, (size_t)Nn);
  count_deg_kernel<<<(Ee + 255) / 256, 256, 0, stream>>>(dst, cnt, Ee);
  inv_clip_kernel<<<(Nn + 255) / 256, 256, 0, stream>>>(cnt, Nn);

  const int EB = (Ee + 63) / 64;
  const int NB = (Nn + 63) / 64;
  const size_t aggN = (size_t)Nn * 64;

  // ---- Encoder ----
  zero_f32_kernel<<<2048, 256, 0, stream>>>(agg, aggN);
  edge_mlp_kernel<5, 4, false, true, false><<<EB, 128, 0, stream>>>(
      x, ea, nullptr, src, dst, w_ee1, enc_e_b1, w_ee2, enc_e_b2, el, agg, Ee);
  node_mlp_kernel<5, 4, false><<<NB, 128, 0, stream>>>(
      x, agg, cnt, w_en1, enc_n_b1, w_en2, enc_n_b2, xl, Hh, Nn);

  // ---- Processor rounds (shared weights, residual updates) ----
  for (int r = 0; r < 2; ++r) {
    zero_f32_kernel<<<2048, 256, 0, stream>>>(agg, aggN);
    edge_mlp_kernel<64, 64, true, true, true><<<EB, 128, 0, stream>>>(
        xl, nullptr, el, src, dst, w_pe1, proc_e_b1, w_pe2, proc_e_b2, el, agg, Ee);
    node_mlp_kernel<64, 4, true><<<NB, 128, 0, stream>>>(
        xl, agg, cnt, w_pn1, proc_n_b1, w_pn2, proc_n_b2, xl, Hh, Nn);
  }

  // ---- Decoder ----
  zero_f32_kernel<<<2048, 256, 0, stream>>>(agg, aggN);
  edge_mlp_kernel<64, 64, true, false, false><<<EB, 128, 0, stream>>>(
      xl, nullptr, el, src, dst, w_de1, dec_e_b1, w_de2, dec_e_b2, el, agg, Ee);
  node_mlp_kernel<64, 1, false><<<NB, 128, 0, stream>>>(
      xl, agg, cnt, w_dn1, dec_n_b1, w_dn2, dec_n_b2, (float*)d_out, OUTD, Nn);

  (void)in_sizes; (void)n_in; (void)out_size; (void)ws_size;
}
